// AttLayer_54528904790879
// MI455X (gfx1250) — matrix operand order repaired and checked by reading, not compiled, not run on hardware
//
#include <hip/hip_runtime.h>
#include <hip/hip_bf16.h>
#include <math.h>

typedef __attribute__((ext_vector_type(16))) __bf16 v16bf;
typedef __attribute__((ext_vector_type(8)))  float  v8f;

#define BB 64
#define SS 2048
#define DD 512
#define AA 256

// Branchless tanh: native CDNA5 v_tanh_f32 when available, otherwise a short
// v_exp/v_rcp sequence (no exec-mask divergence like libm tanhf).
__device__ __forceinline__ float fast_tanh(float x) {
#if __has_builtin(__builtin_amdgcn_tanhf)
    return __builtin_amdgcn_tanhf(x);
#else
    float e = __expf(2.0f * x);
    return 1.0f - 2.0f / (e + 1.0f);
#endif
}

// ---------------------------------------------------------------------------
// Kernel 0: repack W [D=512, A=256] f32 row-major -> bf16 WMMA B-fragment
// layout. Fragment (kk, nt) holds B-tile K=[kk*32,kk*32+32) x N=[nt*16,nt*16+16).
// B layout (16-bit, 32x16): lane half h holds K={8h..8h+7} U {16+8h..16+8h+7}
// (matching A); N = lane%16; 16 bf16 per lane stored contiguously (32B).
// flat index = (((kk*16 + nt)*32 + lane)*16 + elem)
// ---------------------------------------------------------------------------
__global__ void __launch_bounds__(256)
pack_W_bf16(const float* __restrict__ W, __bf16* __restrict__ Wp) {
    int idx  = blockIdx.x * 256 + threadIdx.x;     // 0 .. 512*256-1
    int elem = idx & 15;
    int lane = (idx >> 4) & 31;
    int nt   = (idx >> 9) & 15;
    int kk   = idx >> 13;
    int K = kk * 32 + (lane >> 4) * 8 + (elem & 7) + (elem >> 3) * 16;
    int N = nt * 16 + (lane & 15);
    Wp[idx] = (__bf16)W[K * AA + N];
}

// ---------------------------------------------------------------------------
// Kernel 1: fused GEMM(x,W) + bias + tanh + dot(u) + exp + mask -> ait[B*S]
// Grid: 8192 WGs (16 rows of the flattened [B*S] each), 128 threads = 4 waves.
// Wave w computes N-range [w*64, w*64+64) as 4 WMMA n-tiles.
// ---------------------------------------------------------------------------
__global__ void __launch_bounds__(128)
attn_score(const float* __restrict__ x, const int* __restrict__ mask,
           const __bf16* __restrict__ Wp, const float* __restrict__ bias,
           const float* __restrict__ u, float* __restrict__ ait) {
    const int lane = threadIdx.x & 31;
    const int wave = threadIdx.x >> 5;      // 0..3
    const int m    = lane & 15;             // row / column selector
    const int half = lane >> 4;             // 0 or 1
    const int row0 = blockIdx.x * 16;       // base row in [B*S]

    // per-lane bias / u values for this wave's 4 n-tiles
    const int nbase = wave * 64;
    float bval[4], uval[4];
#pragma unroll
    for (int nt = 0; nt < 4; ++nt) {
        int N = nbase + nt * 16 + m;
        bval[nt] = bias[N];
        uval[nt] = u[N];
    }

    v8f acc[4] = {};

    // A-operand: lane holds row M = m, K elems {h*8..h*8+7} U {16+h*8..16+h*8+7}
    const float* xrow = x + (size_t)(row0 + m) * DD;
    const v16bf* Wf   = (const v16bf*)Wp;

    for (int kk = 0; kk < 16; ++kk) {
        const float* p0 = xrow + kk * 32 + half * 8;
        float tmp[16];
        *(float4*)(tmp + 0)  = *(const float4*)(p0 + 0);
        *(float4*)(tmp + 4)  = *(const float4*)(p0 + 4);
        *(float4*)(tmp + 8)  = *(const float4*)(p0 + 16);
        *(float4*)(tmp + 12) = *(const float4*)(p0 + 20);
        v16bf a;
#pragma unroll
        for (int i = 0; i < 16; ++i) a[i] = (__bf16)tmp[i];

#pragma unroll
        for (int nt = 0; nt < 4; ++nt) {
            v16bf bfrag = Wf[(kk * 16 + (nbase >> 4) + nt) * 32 + lane];
            acc[nt] = __builtin_amdgcn_wmma_f32_16x16x32_bf16(
                false, a, false, bfrag, (short)0, acc[nt], false, false);
        }
    }

    // Epilogue: C layout -> lane holds N = m (+16*nt), rows M = r + 8*half.
    // ait_row = sum_N tanh(c + b[N]) * u[N]
    float partial[8];
#pragma unroll
    for (int r = 0; r < 8; ++r) {
        float p = 0.f;
#pragma unroll
        for (int nt = 0; nt < 4; ++nt) {
            float t = fast_tanh(acc[nt][r] + bval[nt]);
            p = fmaf(t, uval[nt], p);
        }
        partial[r] = p;
    }
    // reduce over N: butterfly within each 16-lane half (all lanes in a half
    // share the same set of 8 rows)
#pragma unroll
    for (int r = 0; r < 8; ++r) {
        float p = partial[r];
        p += __shfl_xor(p, 1, 16);
        p += __shfl_xor(p, 2, 16);
        p += __shfl_xor(p, 4, 16);
        p += __shfl_xor(p, 8, 16);
        partial[r] = p;
    }

    __shared__ float red[4][16];
    if (m == 0) {
#pragma unroll
        for (int r = 0; r < 8; ++r) red[wave][half * 8 + r] = partial[r];
    }
    __syncthreads();

    if (threadIdx.x < 16) {
        int t = threadIdx.x;
        float s = red[0][t] + red[1][t] + red[2][t] + red[3][t];
        int grow = row0 + t;                 // flat (b*S + s) index
        float mk = (mask[grow] != 0) ? 1.0f : 0.0f;
        ait[grow] = __expf(s) * mk;
    }
}

// ---------------------------------------------------------------------------
// Kernel 2: per-batch normalization of ait (in place). One WG per batch.
// ---------------------------------------------------------------------------
__global__ void __launch_bounds__(256)
normalize_ait(float* __restrict__ ait) {
    float* p = ait + (size_t)blockIdx.x * SS;
    float s = 0.f;
    for (int i = threadIdx.x; i < SS; i += 256) s += p[i];
    __shared__ float sh[256];
    sh[threadIdx.x] = s;
    __syncthreads();
    for (int off = 128; off > 0; off >>= 1) {
        if (threadIdx.x < off) sh[threadIdx.x] += sh[threadIdx.x + off];
        __syncthreads();
    }
    float inv = 1.0f / (sh[0] + 1e-7f);
    for (int i = threadIdx.x; i < SS; i += 256) p[i] *= inv;
}

// ---------------------------------------------------------------------------
// Kernel 3: out[b,d] = sum_s x[b,s,d] * w[b,s].  Grid = B*4 (d-chunks of 128),
// 128 threads, one f32 per thread, streaming x once more. w[b,s] is uniform
// across the block -> scalarized load. Deterministic (no atomics).
// ---------------------------------------------------------------------------
__global__ void __launch_bounds__(128)
weighted_sum(const float* __restrict__ x, const float* __restrict__ w,
             float* __restrict__ out) {
    int b = blockIdx.x >> 2;
    int d = (blockIdx.x & 3) * 128 + threadIdx.x;
    const float* xb = x + (size_t)b * SS * DD + d;
    const float* wb = w + (size_t)b * SS;
    float acc = 0.f;
#pragma unroll 4
    for (int s = 0; s < SS; ++s)
        acc = fmaf(xb[(size_t)s * DD], wb[s], acc);
    out[b * DD + d] = acc;
}

// ---------------------------------------------------------------------------
extern "C" void kernel_launch(void* const* d_in, const int* in_sizes, int n_in,
                              void* d_out, int out_size, void* d_ws, size_t ws_size,
                              hipStream_t stream) {
    const float* x    = (const float*)d_in[0];   // [B,S,D]
    const int*   mask = (const int*)d_in[1];     // [B,S]
    const float* W    = (const float*)d_in[2];   // [D,A]
    const float* bias = (const float*)d_in[3];   // [A]
    const float* u    = (const float*)d_in[4];   // [A,1]
    float* out = (float*)d_out;                  // [B,D]

    // workspace layout
    __bf16* Wp  = (__bf16*)d_ws;                              // 512*256*2 = 256KB
    float*  ait = (float*)((char*)d_ws + DD * AA * 2);        // B*S*4     = 512KB

    pack_W_bf16 <<<(DD * AA) / 256, 256, 0, stream>>>(W, Wp);
    attn_score  <<<(BB * SS) / 16, 128, 0, stream>>>(x, mask, Wp, bias, u, ait);
    normalize_ait<<<BB, 256, 0, stream>>>(ait);
    weighted_sum<<<BB * 4, 128, 0, stream>>>(x, ait, out);
}
